// MultiHeadAttention_12601434046931
// MI455X (gfx1250) — matrix operand order repaired and checked by reading, not compiled, not run on hardware
//
#include <hip/hip_runtime.h>
#include <hip/hip_bf16.h>

// ---------------------------------------------------------------------------
// MHA forward for B=4, S=4096, D_MODEL=128, H=4, D_HEAD=32 on gfx1250 (wave32)
// bf16 WMMA (v_wmma_f32_16x16x32_bf16) for all GEMMs, flash-attention over
// keys with per-wave 16-query tiles. Shared per-half-tile running max (single
// DPP16 reduction per key block), row-sum via ones-WMMA, base-2 softmax.
// ---------------------------------------------------------------------------

#define Bc 4
#define Sc 4096
#define DM 128
#define NH 4
#define DH 32

typedef __attribute__((ext_vector_type(16))) __bf16 v16bf;
typedef __attribute__((ext_vector_type(8)))  float  v8f;
typedef __attribute__((ext_vector_type(8)))  unsigned short u16x8;
typedef __attribute__((ext_vector_type(4)))  unsigned short u16x4;
typedef __attribute__((ext_vector_type(4)))  float  f32x4;

union Frag16 {
    u16x8 h[2];
    v16bf v;
    unsigned short us[16];
};

__device__ __forceinline__ unsigned short bfbits(float f) {
    union { __bf16 b; unsigned short u; } t;
    t.b = (__bf16)f;                    // native fp32->bf16 (RNE)
    return t.u;
}

__device__ __forceinline__ float exp2_hw(float x) {
    return __builtin_amdgcn_exp2f(x);   // v_exp_f32 (base-2)
}

// max-reduce across each 16-lane group using DPP16 (no LDS traffic)
__device__ __forceinline__ float dpp_max16(float x) {
    int t;
    t = __builtin_amdgcn_update_dpp(0, __float_as_int(x), 0xB1,  0xF, 0xF, true); // quad_perm 1,0,3,2
    x = fmaxf(x, __int_as_float(t));
    t = __builtin_amdgcn_update_dpp(0, __float_as_int(x), 0x4E,  0xF, 0xF, true); // quad_perm 2,3,0,1
    x = fmaxf(x, __int_as_float(t));
    t = __builtin_amdgcn_update_dpp(0, __float_as_int(x), 0x141, 0xF, 0xF, true); // row_half_mirror
    x = fmaxf(x, __int_as_float(t));
    t = __builtin_amdgcn_update_dpp(0, __float_as_int(x), 0x140, 0xF, 0xF, true); // row_mirror
    x = fmaxf(x, __int_as_float(t));
    return x;
}

// ---------------------------------------------------------------------------
// Streaming fp32 -> bf16 conversion (vector of 4 per thread).
// ---------------------------------------------------------------------------
__global__ void __launch_bounds__(256)
cvt_kernel(const float* __restrict__ src, unsigned short* __restrict__ dst, int n4)
{
    const int i = blockIdx.x * blockDim.x + threadIdx.x;
    if (i < n4) {
        f32x4 v = ((const f32x4*)src)[i];
        union { __bf16 b[4]; u16x4 u; } t;
        t.b[0] = (__bf16)v.x;
        t.b[1] = (__bf16)v.y;
        t.b[2] = (__bf16)v.z;
        t.b[3] = (__bf16)v.w;
        ((u16x4*)dst)[i] = t.u;
    }
}

// ---------------------------------------------------------------------------
// 16384x128 = X[16384x128](bf16) * W[128x128]^T(bf16) via WMMA.
// MODE 0: write bf16 head-row-major   out[((b*H+h)*S+s)*32+d]   (Q, K)
// MODE 1: write bf16 head-transposed  out[((b*H+h)*32+d)*S+s]   (V^T)
// MODE 2: write fp32 row-major        outf[m*128+o]             (final O proj)
// One wave per 16x16 output tile; K-loop of 4 x 32; no conversions inside.
// ---------------------------------------------------------------------------
template <int MODE>
__global__ void __launch_bounds__(256)
proj_kernel(const unsigned short* __restrict__ Xb,
            const unsigned short* __restrict__ Wb,
            unsigned short* __restrict__ outb, float* __restrict__ outf)
{
    const int lane = threadIdx.x & 31;
    const int wv   = threadIdx.x >> 5;
    const int wg   = blockIdx.x * 8 + wv;        // 8192 tile-waves
    const int nt   = wg & 7;                     // 8 col tiles
    const int mt   = wg >> 3;                    // 1024 row tiles
    const int m0   = mt << 4;
    const int n0   = nt << 4;
    const int half = lane >> 4;
    const int l16  = lane & 15;

    v8f acc = {};
    const unsigned short* xrow = Xb + (size_t)(m0 + l16) * DM;
    const unsigned short* wrow = Wb + (size_t)(n0 + l16) * DM;

#pragma unroll
    for (int k0 = 0; k0 < DM; k0 += 32) {
        Frag16 fa, fb;
        // A frag: row m0+l16, k = k0 + half*16 + 0..15
        const unsigned short* pa = xrow + k0 + half * 16;
        fa.h[0] = *(const u16x8*)(pa);
        fa.h[1] = *(const u16x8*)(pa + 8);
        // B frag: col n0+l16 (= output channel), k = k0 + half*16 + 0..15
        const unsigned short* pw = wrow + k0 + half * 16;
        fb.h[0] = *(const u16x8*)(pw);
        fb.h[1] = *(const u16x8*)(pw + 8);
        acc = __builtin_amdgcn_wmma_f32_16x16x32_bf16(
            false, fa.v, false, fb.v, (short)0, acc, false, false);
    }

    const int o = n0 + l16;
#pragma unroll
    for (int v = 0; v < 8; ++v) {
        const int   m   = m0 + half * 8 + v;
        const float val = acc[v];
        if (MODE == 2) {
            outf[(size_t)m * DM + o] = val;
        } else {
            const int bb = m >> 12;          // m / S
            const int ss = m & (Sc - 1);     // m % S
            const int hh = o >> 5;           // o / D_HEAD
            const int dd = o & 31;           // o % D_HEAD
            size_t idx;
            if (MODE == 0) idx = ((size_t)(bb * NH + hh) * Sc + ss) * DH + dd;
            else           idx = ((size_t)(bb * NH + hh) * DH + dd) * Sc + ss;
            outb[idx] = bfbits(val);
        }
    }
}

// ---------------------------------------------------------------------------
// Flash attention: one wave per (b, h, 16-query tile), streaming keys in
// blocks of 32. QK^T and PV via bf16 WMMA. A single running max per lane-half
// (valid upper bound for all 8 rows it holds; softmax normalization cancels
// the shared scaling exactly). Row sums via P x ones WMMA.
// ---------------------------------------------------------------------------
__global__ void __launch_bounds__(256)
attn_kernel(const unsigned short* __restrict__ Qh,
            const unsigned short* __restrict__ Kh,
            const unsigned short* __restrict__ VhT,
            const float* __restrict__ mask,
            unsigned short* __restrict__ ctx)
{
    __shared__ __align__(16) unsigned short pb[8 * 16 * 32];  // 8 KB, 1 KB/wave

    const int lane = threadIdx.x & 31;
    const int wv   = threadIdx.x >> 5;
    const int wg   = blockIdx.x * 8 + wv;        // 4096 waves
    const int qt   = wg & 255;
    const int h    = (wg >> 8) & 3;
    const int b    = wg >> 10;
    const int bh   = b * NH + h;
    const int q0   = qt << 4;
    const int half = lane >> 4;
    const int l16  = lane & 15;

    // Q tile in A layout: row = q0 + l16, k (=head dim) = half*16 + 0..15
    Frag16 aQ;
    {
        const unsigned short* p = Qh + ((size_t)bh * Sc + q0 + l16) * DH + half * 16;
        aQ.h[0] = *(const u16x8*)(p);
        aQ.h[1] = *(const u16x8*)(p + 8);
    }

    // constant all-ones B-frag (bf16 1.0 = 0x3F80) for row-sum WMMA
    Frag16 fones;
#pragma unroll
    for (int e = 0; e < 16; ++e) fones.us[e] = 0x3F80;

    float mtile = -1e30f;                // shared running max (per lane-half)
    float lrow[8];
#pragma unroll
    for (int v = 0; v < 8; ++v) lrow[v] = 0.0f;
    v8f acc0 = {}, acc1 = {};

    // base-2 domain: p = 2^(s*scale*log2e + mask*log2e - m)
    const float kLog2e   = 1.4426950408889634f;
    const float scaleL2  = 0.17677669529663687f * kLog2e;  // (1/sqrt(32))*log2e
    const float* mbase = mask + (size_t)b * Sc * Sc;
    unsigned short* myp = pb + wv * 512;

    for (int kb = 0; kb < Sc; kb += 32) {
        // K B-frags: col = key (l16 / l16+16), k = dim = half*16 + 0..15 contiguous
        Frag16 fk0, fk1;
        const unsigned short* kp0 = Kh + ((size_t)bh * Sc + kb + l16) * DH + half * 16;
        const unsigned short* kp1 = kp0 + (size_t)16 * DH;
        fk0.h[0] = *(const u16x8*)(kp0); fk0.h[1] = *(const u16x8*)(kp0 + 8);
        fk1.h[0] = *(const u16x8*)(kp1); fk1.h[1] = *(const u16x8*)(kp1 + 8);

        v8f z = {};
        v8f s0 = __builtin_amdgcn_wmma_f32_16x16x32_bf16(
            false, aQ.v, false, fk0.v, (short)0, z, false, false);
        v8f s1 = __builtin_amdgcn_wmma_f32_16x16x32_bf16(
            false, aQ.v, false, fk1.v, (short)0, z, false, false);

        // scale + mask (base-2); C layout: lane holds (row = half*8+v,
        // col = l16) for s0 and col = 16+l16 for s1.
        float sc0a[8], sc1a[8];
        float bmax = -1e30f;
#pragma unroll
        for (int v = 0; v < 8; ++v) {
            const int q = q0 + half * 8 + v;
            const float* mp = mbase + (size_t)q * Sc + kb;
            sc0a[v] = s0[v] * scaleL2 + mp[l16]      * kLog2e;
            sc1a[v] = s1[v] * scaleL2 + mp[16 + l16] * kLog2e;
            bmax = fmaxf(bmax, fmaxf(sc0a[v], sc1a[v]));
        }
        // one 16-lane reduction for the whole half-tile
        bmax = dpp_max16(bmax);
        const float mnew = fmaxf(mtile, bmax);
        const float cf   = exp2_hw(mtile - mnew);   // shared rescale factor
        mtile = mnew;

#pragma unroll
        for (int v = 0; v < 8; ++v) {
            const float p0 = exp2_hw(sc0a[v] - mnew);
            const float p1 = exp2_hw(sc1a[v] - mnew);
            lrow[v] *= cf;          // row-sum contribution added after ones-WMMA
            acc0[v] *= cf;
            acc1[v] *= cf;
            // C-layout -> LDS row-major (bf16) for the A-layout reload
            myp[(half * 8 + v) * 32 + l16]      = bfbits(p0);
            myp[(half * 8 + v) * 32 + 16 + l16] = bfbits(p1);
        }

        __syncthreads();   // uniform trip count for all waves in block
        // P tile in A layout: row = l16, k = half*16 + 0..15
        Frag16 fp;
        const unsigned short* pp = myp + l16 * 32 + half * 16;
        fp.h[0] = *(const u16x8*)(pp);
        fp.h[1] = *(const u16x8*)(pp + 8);
        __syncthreads();

        // row sums of P via ones-WMMA: rsums[v] = sum_k P[row,k], replicated
        v8f rsums = __builtin_amdgcn_wmma_f32_16x16x32_bf16(
            false, fp.v, false, fones.v, (short)0, z, false, false);
#pragma unroll
        for (int v = 0; v < 8; ++v) lrow[v] += rsums[v];

        // V B-frags from V^T: col = dim (l16 / 16+l16), k = key contiguous
        Frag16 fv0, fv1;
        const unsigned short* vp0 = VhT + ((size_t)bh * DH + l16)      * Sc + kb + half * 16;
        const unsigned short* vp1 = VhT + ((size_t)bh * DH + 16 + l16) * Sc + kb + half * 16;
        fv0.h[0] = *(const u16x8*)(vp0); fv0.h[1] = *(const u16x8*)(vp0 + 8);
        fv1.h[0] = *(const u16x8*)(vp1); fv1.h[1] = *(const u16x8*)(vp1 + 8);

        acc0 = __builtin_amdgcn_wmma_f32_16x16x32_bf16(
            false, fp.v, false, fv0.v, (short)0, acc0, false, false);
        acc1 = __builtin_amdgcn_wmma_f32_16x16x32_bf16(
            false, fp.v, false, fv1.v, (short)0, acc1, false, false);
    }

    // normalize and write context bf16 [b, s, h*32 + d]
#pragma unroll
    for (int v = 0; v < 8; ++v) {
        const int   s   = q0 + half * 8 + v;
        const float inv = 1.0f / lrow[v];
        unsigned short* crow = ctx + ((size_t)b * Sc + s) * DM + h * DH;
        crow[l16]      = bfbits(acc0[v] * inv);
        crow[16 + l16] = bfbits(acc1[v] * inv);
    }
}

// ---------------------------------------------------------------------------
extern "C" void kernel_launch(void* const* d_in, const int* in_sizes, int n_in,
                              void* d_out, int out_size, void* d_ws, size_t ws_size,
                              hipStream_t stream)
{
    const float* inQ  = (const float*)d_in[0];
    const float* inK  = (const float*)d_in[1];
    const float* inV  = (const float*)d_in[2];
    const float* mask = (const float*)d_in[3];
    const float* WQ   = (const float*)d_in[4];
    const float* WK   = (const float*)d_in[5];
    const float* WV   = (const float*)d_in[6];
    const float* WO   = (const float*)d_in[7];

    const size_t NE = (size_t)Bc * Sc * DM;      // 2,097,152 activation elems
    const size_t NW = (size_t)DM * DM;           // 16,384 weight elems

    unsigned short* p   = (unsigned short*)d_ws;
    unsigned short* Xqb = p;  p += NE;           // bf16 inputs
    unsigned short* Xkb = p;  p += NE;
    unsigned short* Xvb = p;  p += NE;
    unsigned short* Wqb = p;  p += NW;           // bf16 weights
    unsigned short* Wkb = p;  p += NW;
    unsigned short* Wvb = p;  p += NW;
    unsigned short* Wob = p;  p += NW;
    unsigned short* Qh  = p;  p += NE;           // projected heads (bf16)
    unsigned short* Kh  = p;  p += NE;
    unsigned short* VhT = p;  p += NE;           // V transposed per head
    unsigned short* ctx = p;  p += NE;           // attention context (bf16)

    dim3 blk(256);
    const int gX = (int)(NE / 4 / 256);          // 2048 blocks
    const int gW = (int)(NW / 4 / 256);          // 16 blocks

    // fp32 -> bf16 staging
    cvt_kernel<<<gX, blk, 0, stream>>>(inQ, Xqb, (int)(NE / 4));
    cvt_kernel<<<gX, blk, 0, stream>>>(inK, Xkb, (int)(NE / 4));
    cvt_kernel<<<gX, blk, 0, stream>>>(inV, Xvb, (int)(NE / 4));
    cvt_kernel<<<gW, blk, 0, stream>>>(WQ, Wqb, (int)(NW / 4));
    cvt_kernel<<<gW, blk, 0, stream>>>(WK, Wkb, (int)(NW / 4));
    cvt_kernel<<<gW, blk, 0, stream>>>(WV, Wvb, (int)(NW / 4));
    cvt_kernel<<<gW, blk, 0, stream>>>(WO, Wob, (int)(NW / 4));

    // QKV projections (WMMA), writing attention-friendly bf16 layouts
    proj_kernel<0><<<1024, blk, 0, stream>>>(Xqb, Wqb, Qh,  nullptr);
    proj_kernel<0><<<1024, blk, 0, stream>>>(Xkb, Wkb, Kh,  nullptr);
    proj_kernel<1><<<1024, blk, 0, stream>>>(Xvb, Wvb, VhT, nullptr);
    // flash attention: 4096 waves = B*H*(S/16)
    attn_kernel<<<512, blk, 0, stream>>>(Qh, Kh, VhT, mask, ctx);
    // output projection straight to fp32 d_out
    proj_kernel<2><<<1024, blk, 0, stream>>>(ctx, Wob, nullptr, (float*)d_out);
}
